// model_82695300317474
// MI455X (gfx1250) — compile-verified, matrix operand order checked by reading, not run on hardware
//
#include <hip/hip_runtime.h>
#include <math.h>

// Model dims
#define S_ 128
#define B_ 16
#define TV_ 20
#define T_ 64
#define H_ 512
#define Hh_ 256
#define V_ 32000
#define VV_ 5000
#define BOS_ 3

typedef __bf16 bf16_t;
typedef __attribute__((ext_vector_type(16))) __bf16 v16bf;
typedef __attribute__((ext_vector_type(8))) float v8f;

__device__ __forceinline__ float sigmf(float x) { return 1.0f / (1.0f + __expf(-x)); }

// Wave index, provably scalar so strip-mine loops use s_cbranch (no exec
// masking around WMMA, which requires EXEC all-ones and otherwise spills accs).
__device__ __forceinline__ int wave_id() {
  return __builtin_amdgcn_readfirstlane((int)(threadIdx.x >> 5));
}

// Load one 16x32 bf16 WMMA fragment (A layout; B^T layout is identical when the
// weight matrix is stored N x K row-major). Lane L (<16): row/col = L, K chunk
// base 0; lane L>=16: same row, K chunk base +8; each half also covers K+16.
__device__ __forceinline__ v16bf ld_frag(const bf16_t* base, int ld) {
  const int lane = threadIdx.x & 31;
  const bf16_t* p = base + (size_t)(lane & 15) * ld + ((lane >> 4) << 3);
  v16bf v;
  ((float4*)&v)[0] = ((const float4*)p)[0];
  ((float4*)&v)[1] = ((const float4*)(p + 16))[0];
  return v;
}

__device__ __forceinline__ void wmma_acc(v8f& acc, v16bf a, v16bf b) {
  acc = __builtin_amdgcn_wmma_f32_16x16x32_bf16(false, a, false, b, (short)0, acc,
                                                false, false);
}

// Workgroup (8 waves) GEMM: G[16 x N] = A1(16xK1) @ W1^T (+ A2(16xK2) @ W2^T)
// W stored N x K row-major. Result written to LDS gbuf (f32, row stride N).
__device__ void wg_gemm_to_lds(float* gbuf, int N,
                               const bf16_t* A1, const bf16_t* W1, int K1,
                               const bf16_t* A2, const bf16_t* W2, int K2) {
  const int wave = wave_id();
  const int lane = threadIdx.x & 31;
  const int ntiles = N >> 4;
  for (int nt = wave; nt < ntiles; nt += 8) {
    v8f acc = {0.f, 0.f, 0.f, 0.f, 0.f, 0.f, 0.f, 0.f};
    for (int k = 0; k < K1; k += 32) {
      v16bf a = ld_frag(A1 + k, K1);
      v16bf b = ld_frag(W1 + (size_t)nt * 16 * K1 + k, K1);
      wmma_acc(acc, a, b);
    }
    if (A2) {  // scalar (pointer nullness) branch
      for (int k = 0; k < K2; k += 32) {
        v16bf a = ld_frag(A2 + k, K2);
        v16bf b = ld_frag(W2 + (size_t)nt * 16 * K2 + k, K2);
        wmma_acc(acc, a, b);
      }
    }
    const int n = nt * 16 + (lane & 15);
    const int mb = (lane >> 4) * 8;
#pragma unroll
    for (int i = 0; i < 8; ++i) gbuf[(size_t)(mb + i) * N + n] = acc[i];
  }
}

// ---------------------------------------------------------------------------
// Elementwise / conversion kernels
// ---------------------------------------------------------------------------
__global__ void cvt_kernel(const float* __restrict__ src, bf16_t* __restrict__ dst,
                           size_t n) {
  for (size_t i = (size_t)blockIdx.x * 256 + threadIdx.x; i < n;
       i += (size_t)gridDim.x * 256)
    dst[i] = (bf16_t)src[i];
}

__global__ void zero_kernel(float* __restrict__ p, size_t n) {
  for (size_t i = (size_t)blockIdx.x * 256 + threadIdx.x; i < n;
       i += (size_t)gridDim.x * 256)
    p[i] = 0.f;
}

__global__ void zero_bf16_kernel(bf16_t* __restrict__ p, size_t n) {
  for (size_t i = (size_t)blockIdx.x * 256 + threadIdx.x; i < n;
       i += (size_t)gridDim.x * 256)
    p[i] = (bf16_t)0.f;
}

// Gather encoder embeddings into bf16 activation matrix, rows r = s*16+b.
__global__ void enc_embed_kernel(const float* __restrict__ emb,
                                 const int* __restrict__ inp,
                                 bf16_t* __restrict__ out) {
  const size_t total = (size_t)S_ * B_ * H_;
  for (size_t idx = (size_t)blockIdx.x * 256 + threadIdx.x; idx < total;
       idx += (size_t)gridDim.x * 256) {
    const size_t row = idx >> 9;
    const int j = (int)(idx & 511);
    const int s = (int)(row >> 4), b = (int)(row & 15);
    const int tok = inp[b * S_ + s];
    out[idx] = (bf16_t)emb[(size_t)tok * H_ + j];
  }
}

// ---------------------------------------------------------------------------
// Generic big WMMA GEMM: C = A(MxK bf16) @ W^T (W is NxK bf16) + bias1 + bias2
// Rows r = t*16+b; stored C[b*sB + t*sT + n] (handles logits layout directly).
// Each wave computes a 16x64 strip (1 A-frag reused over 4 B-frags).
// W MUST be padded to nblocks*512 rows (pad rows zeroed); no bounds checks in
// the K-loop so accumulators stay in VGPRs (no exec-masked WMMA).
// ---------------------------------------------------------------------------
__global__ void gemm_bf16_kernel(const bf16_t* __restrict__ A,
                                 const bf16_t* __restrict__ W,
                                 const float* __restrict__ bias1,
                                 const float* __restrict__ bias2,
                                 float* __restrict__ C, int M, int N, int K,
                                 long sB, long sT) {
  const int wave = wave_id();
  const int lane = threadIdx.x & 31;
  const int NT = 4;
  const int ntilesTot = (N + 15) >> 4;
  const int nblocks = (ntilesTot + 8 * NT - 1) / (8 * NT);
  const int mt = blockIdx.x / nblocks;
  const int nb = blockIdx.x % nblocks;
  const int nt0 = nb * (8 * NT) + wave * NT;
  v8f acc[NT];
#pragma unroll
  for (int j = 0; j < NT; ++j) acc[j] = (v8f){0.f, 0.f, 0.f, 0.f, 0.f, 0.f, 0.f, 0.f};
  const bf16_t* Ab = A + (size_t)mt * 16 * K;
  for (int k = 0; k < K; k += 32) {
    v16bf a = ld_frag(Ab + k, K);
#pragma unroll
    for (int j = 0; j < NT; ++j) {
      v16bf b = ld_frag(W + (size_t)(nt0 + j) * 16 * K + k, K);
      wmma_acc(acc[j], a, b);
    }
  }
  const int nlo = lane & 15;
  const int mb = (lane >> 4) * 8;
  for (int j = 0; j < NT; ++j) {
    const int n = (nt0 + j) * 16 + nlo;
    if (n < N) {  // epilogue-only mask (outside K-loop)
      const float badd = (bias1 ? bias1[n] : 0.f) + (bias2 ? bias2[n] : 0.f);
#pragma unroll
      for (int i = 0; i < 8; ++i) {
        const int m = mt * 16 + mb + i;
        C[(size_t)(m & 15) * sB + (size_t)(m >> 4) * sT + n] = acc[j][i] + badd;
      }
    }
  }
}

// ---------------------------------------------------------------------------
// Encoder recurrent kernel. grid.x = 2 (direction); one WG per direction loops
// over all 128 timesteps. xpart already holds x@Wih^T + b_ih + b_hh.
// ---------------------------------------------------------------------------
__global__ void enc_lstm_kernel(const float* __restrict__ xpart,   // [2][S*B][1024]
                                const bf16_t* __restrict__ whh,    // [2][1024][256]
                                float* __restrict__ hs_f,          // [S*B][512] or null
                                bf16_t* __restrict__ hs_bf,        // [S*B][512] or null
                                float* __restrict__ hfin,          // [B][512]
                                float* __restrict__ cfin) {
  const int d = blockIdx.x;
  const float* xp = xpart + (size_t)d * S_ * B_ * 1024;
  const bf16_t* W = whh + (size_t)d * 1024 * 256;
  extern __shared__ char smem[];
  float* g = (float*)smem;                      // 16*1024 f32
  float* cbuf = (float*)(smem + 65536);         // 16*256 f32
  bf16_t* hbf = (bf16_t*)(smem + 65536 + 16384);// 16*256 bf16
  for (int i = threadIdx.x; i < 16 * 256; i += 256) {
    cbuf[i] = 0.f;
    hbf[i] = (bf16_t)0.f;
  }
  __syncthreads();
  for (int t = 0; t < S_; ++t) {
    const int s = d ? (S_ - 1 - t) : t;
    wg_gemm_to_lds(g, 1024, hbf, W, 256, nullptr, nullptr, 0);
    __syncthreads();
    for (int e = threadIdx.x; e < 16 * 256; e += 256) {
      const int b = e >> 8, j = e & 255;
      const float* xr = xp + ((size_t)s * 16 + b) * 1024;
      const float* gr = g + (size_t)b * 1024;
      float gi = xr[j] + gr[j];
      float gf = xr[256 + j] + gr[256 + j];
      float gg = xr[512 + j] + gr[512 + j];
      float go = xr[768 + j] + gr[768 + j];
      float c = sigmf(gf) * cbuf[e] + sigmf(gi) * tanhf(gg);
      float h = sigmf(go) * tanhf(c);
      cbuf[e] = c;
      hbf[e] = (bf16_t)h;
      const size_t orow = ((size_t)s * 16 + b) * 512 + (size_t)d * 256 + j;
      if (hs_f) hs_f[orow] = h;
      if (hs_bf) hs_bf[orow] = (bf16_t)h;
      if (t == S_ - 1) {
        hfin[(size_t)b * 512 + d * 256 + j] = h;
        cfin[(size_t)b * 512 + d * 256 + j] = c;
      }
    }
    __syncthreads();
  }
}

// ---------------------------------------------------------------------------
// Verb decoder: single WG, loops over TV steps. Stores decout (venc) and
// tanh(out-proj) history (vopn_bf) for the hoisted logits GEMM.
// ---------------------------------------------------------------------------
__global__ void vdec_kernel(const int* __restrict__ verbs, const float* __restrict__ vemb,
                            const bf16_t* Wih0, const bf16_t* Wih1,
                            const bf16_t* Whh0, const bf16_t* Whh1,
                            const float* __restrict__ bih, const float* __restrict__ bhh,
                            const bf16_t* Wlinin_, const bf16_t* Wlinout_,
                            const float* __restrict__ enc_f,
                            float* __restrict__ h_state, float* __restrict__ c_state,
                            float* __restrict__ op_state,
                            float* __restrict__ venc_f, bf16_t* __restrict__ opn_bf) {
  extern __shared__ char smem[];
  float* g = (float*)smem;                              // 16*2048 f32
  bf16_t* xbf = (bf16_t*)(smem + 131072);               // 16*1024
  bf16_t* h0bf = (bf16_t*)(smem + 131072 + 32768);      // 16*512
  bf16_t* h1bf = (bf16_t*)(smem + 131072 + 49152);      // 16*512
  float* att = (float*)(smem + 131072 + 65536);         // 16*128
  float* q = g;                                         // overlays on g (post-gates)
  float* cc = g + 16 * 512;
  float* opn = g + 16 * 1024;
  const int tid = threadIdx.x;
  for (int t = 0; t < TV_; ++t) {
    __syncthreads();
    for (int e = tid; e < 16 * 512; e += 256) {
      const int b = e >> 9, j = e & 511;
      const int tok = (t == 0) ? BOS_ : verbs[b * TV_ + (t - 1)];
      xbf[b * 1024 + j] = (bf16_t)vemb[(size_t)tok * 512 + j];
      xbf[b * 1024 + 512 + j] = (bf16_t)op_state[e];
      h0bf[e] = (bf16_t)h_state[e];
      h1bf[e] = (bf16_t)h_state[16 * 512 + e];
    }
    __syncthreads();
    wg_gemm_to_lds(g, 2048, xbf, Wih0, 1024, h0bf, Whh0, 512);
    __syncthreads();
    for (int e = tid; e < 16 * 512; e += 256) {
      const int b = e >> 9, j = e & 511;
      const float* gr = g + (size_t)b * 2048;
      float gi = gr[j] + bih[j] + bhh[j];
      float gf = gr[512 + j] + bih[512 + j] + bhh[512 + j];
      float gg = gr[1024 + j] + bih[1024 + j] + bhh[1024 + j];
      float go = gr[1536 + j] + bih[1536 + j] + bhh[1536 + j];
      float c = sigmf(gf) * c_state[e] + sigmf(gi) * tanhf(gg);
      float h = sigmf(go) * tanhf(c);
      c_state[e] = c; h_state[e] = h; h0bf[e] = (bf16_t)h;
    }
    __syncthreads();
    wg_gemm_to_lds(g, 2048, h0bf, Wih1, 512, h1bf, Whh1, 512);
    __syncthreads();
    for (int e = tid; e < 16 * 512; e += 256) {
      const int b = e >> 9, j = e & 511;
      const float* gr = g + (size_t)b * 2048;
      const float* bi = bih + 2048;
      const float* bh2 = bhh + 2048;
      float gi = gr[j] + bi[j] + bh2[j];
      float gf = gr[512 + j] + bi[512 + j] + bh2[512 + j];
      float gg = gr[1024 + j] + bi[1024 + j] + bh2[1024 + j];
      float go = gr[1536 + j] + bi[1536 + j] + bh2[1536 + j];
      float c = sigmf(gf) * c_state[16 * 512 + e] + sigmf(gi) * tanhf(gg);
      float h = sigmf(go) * tanhf(c);
      c_state[16 * 512 + e] = c; h_state[16 * 512 + e] = h;
      h1bf[e] = (bf16_t)h;
      venc_f[((size_t)t * 16 + b) * 512 + j] = h;
    }
    __syncthreads();
    wg_gemm_to_lds(q, 512, h1bf, Wlinin_, 512, nullptr, nullptr, 0);
    __syncthreads();
    for (int e = tid; e < 16 * 128; e += 256) {
      const int b = e >> 7, s = e & 127;
      const float* er = enc_f + ((size_t)s * 16 + b) * 512;
      const float* qr = q + (size_t)b * 512;
      float acc = 0.f;
      for (int hh = 0; hh < 512; ++hh) acc += er[hh] * qr[hh];
      att[e] = acc;
    }
    __syncthreads();
    if (tid < 16) {
      float mx = -1e30f;
      for (int s = 0; s < 128; ++s) mx = fmaxf(mx, att[tid * 128 + s]);
      float sum = 0.f;
      for (int s = 0; s < 128; ++s) { float w = __expf(att[tid * 128 + s] - mx); att[tid * 128 + s] = w; sum += w; }
      float inv = 1.f / sum;
      for (int s = 0; s < 128; ++s) att[tid * 128 + s] *= inv;
    }
    __syncthreads();
    for (int e = tid; e < 16 * 512; e += 256) {
      const int b = e >> 9, j = e & 511;
      float acc = 0.f;
      const float* wr = att + b * 128;
      for (int s = 0; s < 128; ++s) acc += wr[s] * enc_f[((size_t)s * 16 + b) * 512 + j];
      cc[e] = acc;
    }
    __syncthreads();
    for (int e = tid; e < 16 * 512; e += 256) {
      const int b = e >> 9, j = e & 511;
      xbf[b * 1024 + j] = (bf16_t)cc[e];
      xbf[b * 1024 + 512 + j] = h1bf[e];
    }
    __syncthreads();
    wg_gemm_to_lds(opn, 512, xbf, Wlinout_, 1024, nullptr, nullptr, 0);
    __syncthreads();
    for (int e = tid; e < 16 * 512; e += 256) {
      const int b = e >> 9, j = e & 511;
      float v = tanhf(opn[(size_t)b * 512 + j]);
      op_state[e] = v;
      opn_bf[((size_t)t * 16 + b) * 512 + j] = (bf16_t)v;
    }
  }
  __syncthreads();
  for (int e = tid; e < 16 * 512; e += 256) op_state[e] = 0.f;  // dstep starts with op=0
}

// ---------------------------------------------------------------------------
// Main decoder: like vdec but with two attentions (enc + venc) and 3-way concat.
// ---------------------------------------------------------------------------
__global__ void dec_kernel(const int* __restrict__ outtok, const float* __restrict__ decemb,
                           const bf16_t* Wih0, const bf16_t* Wih1,
                           const bf16_t* Whh0, const bf16_t* Whh1,
                           const float* __restrict__ bih, const float* __restrict__ bhh,
                           const bf16_t* Wlinin_, const bf16_t* Wdvlinin_,
                           const bf16_t* Wlinout_,
                           const float* __restrict__ enc_f, const float* __restrict__ venc_f,
                           float* __restrict__ h_state, float* __restrict__ c_state,
                           float* __restrict__ op_state, bf16_t* __restrict__ opn_bf) {
  extern __shared__ char smem[];
  float* g = (float*)smem;                                   // 16*2048 f32
  bf16_t* xbf = (bf16_t*)(smem + 131072);                    // 16*1024
  bf16_t* h0bf = (bf16_t*)(smem + 131072 + 32768);           // 16*512
  bf16_t* h1bf = (bf16_t*)(smem + 131072 + 49152);           // 16*512
  bf16_t* cat3 = (bf16_t*)(smem + 131072 + 65536);           // 16*1536
  float* att = (float*)(smem + 131072 + 65536 + 49152);      // 16*128
  float* att2 = (float*)(smem + 131072 + 65536 + 49152 + 8192); // 16*20
  float* q = g;
  float* cc = g + 16 * 512;
  float* vcc = g + 2 * 16 * 512;
  float* opn = g + 3 * 16 * 512;
  const int tid = threadIdx.x;
  for (int t = 0; t < T_; ++t) {
    __syncthreads();
    for (int e = tid; e < 16 * 512; e += 256) {
      const int b = e >> 9, j = e & 511;
      const int tok = (t == 0) ? BOS_ : outtok[b * T_ + (t - 1)];
      xbf[b * 1024 + j] = (bf16_t)decemb[(size_t)tok * 512 + j];
      xbf[b * 1024 + 512 + j] = (bf16_t)op_state[e];
      h0bf[e] = (bf16_t)h_state[e];
      h1bf[e] = (bf16_t)h_state[16 * 512 + e];
    }
    __syncthreads();
    wg_gemm_to_lds(g, 2048, xbf, Wih0, 1024, h0bf, Whh0, 512);
    __syncthreads();
    for (int e = tid; e < 16 * 512; e += 256) {
      const int b = e >> 9, j = e & 511;
      const float* gr = g + (size_t)b * 2048;
      float gi = gr[j] + bih[j] + bhh[j];
      float gf = gr[512 + j] + bih[512 + j] + bhh[512 + j];
      float gg = gr[1024 + j] + bih[1024 + j] + bhh[1024 + j];
      float go = gr[1536 + j] + bih[1536 + j] + bhh[1536 + j];
      float c = sigmf(gf) * c_state[e] + sigmf(gi) * tanhf(gg);
      float h = sigmf(go) * tanhf(c);
      c_state[e] = c; h_state[e] = h; h0bf[e] = (bf16_t)h;
    }
    __syncthreads();
    wg_gemm_to_lds(g, 2048, h0bf, Wih1, 512, h1bf, Whh1, 512);
    __syncthreads();
    for (int e = tid; e < 16 * 512; e += 256) {
      const int b = e >> 9, j = e & 511;
      const float* gr = g + (size_t)b * 2048;
      const float* bi = bih + 2048;
      const float* bh2 = bhh + 2048;
      float gi = gr[j] + bi[j] + bh2[j];
      float gf = gr[512 + j] + bi[512 + j] + bh2[512 + j];
      float gg = gr[1024 + j] + bi[1024 + j] + bh2[1024 + j];
      float go = gr[1536 + j] + bi[1536 + j] + bh2[1536 + j];
      float c = sigmf(gf) * c_state[16 * 512 + e] + sigmf(gi) * tanhf(gg);
      float h = sigmf(go) * tanhf(c);
      c_state[16 * 512 + e] = c; h_state[16 * 512 + e] = h;
      h1bf[e] = (bf16_t)h;
    }
    __syncthreads();
    // attention over encoder states
    wg_gemm_to_lds(q, 512, h1bf, Wlinin_, 512, nullptr, nullptr, 0);
    __syncthreads();
    for (int e = tid; e < 16 * 128; e += 256) {
      const int b = e >> 7, s = e & 127;
      const float* er = enc_f + ((size_t)s * 16 + b) * 512;
      const float* qr = q + (size_t)b * 512;
      float acc = 0.f;
      for (int hh = 0; hh < 512; ++hh) acc += er[hh] * qr[hh];
      att[e] = acc;
    }
    __syncthreads();
    if (tid < 16) {
      float mx = -1e30f;
      for (int s = 0; s < 128; ++s) mx = fmaxf(mx, att[tid * 128 + s]);
      float sum = 0.f;
      for (int s = 0; s < 128; ++s) { float w = __expf(att[tid * 128 + s] - mx); att[tid * 128 + s] = w; sum += w; }
      float inv = 1.f / sum;
      for (int s = 0; s < 128; ++s) att[tid * 128 + s] *= inv;
    }
    __syncthreads();
    for (int e = tid; e < 16 * 512; e += 256) {
      const int b = e >> 9, j = e & 511;
      float acc = 0.f;
      const float* wr = att + b * 128;
      for (int s = 0; s < 128; ++s) acc += wr[s] * enc_f[((size_t)s * 16 + b) * 512 + j];
      cc[e] = acc;
    }
    __syncthreads();
    // attention over verb encodings
    wg_gemm_to_lds(q, 512, h1bf, Wdvlinin_, 512, nullptr, nullptr, 0);
    __syncthreads();
    for (int e = tid; e < 16 * TV_; e += 256) {
      const int b = e / TV_, s = e % TV_;
      const float* er = venc_f + ((size_t)s * 16 + b) * 512;
      const float* qr = q + (size_t)b * 512;
      float acc = 0.f;
      for (int hh = 0; hh < 512; ++hh) acc += er[hh] * qr[hh];
      att2[b * TV_ + s] = acc;
    }
    __syncthreads();
    if (tid < 16) {
      float mx = -1e30f;
      for (int s = 0; s < TV_; ++s) mx = fmaxf(mx, att2[tid * TV_ + s]);
      float sum = 0.f;
      for (int s = 0; s < TV_; ++s) { float w = __expf(att2[tid * TV_ + s] - mx); att2[tid * TV_ + s] = w; sum += w; }
      float inv = 1.f / sum;
      for (int s = 0; s < TV_; ++s) att2[tid * TV_ + s] *= inv;
    }
    __syncthreads();
    for (int e = tid; e < 16 * 512; e += 256) {
      const int b = e >> 9, j = e & 511;
      float acc = 0.f;
      const float* wr = att2 + b * TV_;
      for (int s = 0; s < TV_; ++s) acc += wr[s] * venc_f[((size_t)s * 16 + b) * 512 + j];
      vcc[e] = acc;
    }
    __syncthreads();
    for (int e = tid; e < 16 * 512; e += 256) {
      const int b = e >> 9, j = e & 511;
      cat3[b * 1536 + j] = (bf16_t)cc[e];
      cat3[b * 1536 + 512 + j] = h1bf[e];
      cat3[b * 1536 + 1024 + j] = (bf16_t)vcc[e];
    }
    __syncthreads();
    wg_gemm_to_lds(opn, 512, cat3, Wlinout_, 1536, nullptr, nullptr, 0);
    __syncthreads();
    for (int e = tid; e < 16 * 512; e += 256) {
      const int b = e >> 9, j = e & 511;
      float v = tanhf(opn[(size_t)b * 512 + j]);
      op_state[e] = v;
      opn_bf[((size_t)t * 16 + b) * 512 + j] = (bf16_t)v;
    }
  }
}

// ---------------------------------------------------------------------------
// Host orchestration
// ---------------------------------------------------------------------------
static inline char* bump(char*& p, size_t bytes) {
  char* r = p;
  p += (bytes + 255) & ~(size_t)255;
  return r;
}

extern "C" void kernel_launch(void* const* d_in, const int* in_sizes, int n_in,
                              void* d_out, int out_size, void* d_ws, size_t ws_size,
                              hipStream_t stream) {
  (void)in_sizes; (void)n_in; (void)out_size; (void)ws_size;
  const float* encemb = (const float*)d_in[0];
  const float* decemb = (const float*)d_in[1];
  const float* vemb = (const float*)d_in[2];
  const float* enc_w_ih = (const float*)d_in[3];
  const float* enc_w_hh = (const float*)d_in[4];
  const float* enc_b_ih = (const float*)d_in[5];
  const float* enc_b_hh = (const float*)d_in[6];
  const float* vdec_w_ih0 = (const float*)d_in[7];
  const float* vdec_w_ih1 = (const float*)d_in[8];
  const float* vdec_w_hh = (const float*)d_in[9];
  const float* vdec_b_ih = (const float*)d_in[10];
  const float* vdec_b_hh = (const float*)d_in[11];
  const float* dec_w_ih0 = (const float*)d_in[12];
  const float* dec_w_ih1 = (const float*)d_in[13];
  const float* dec_w_hh = (const float*)d_in[14];
  const float* dec_b_ih = (const float*)d_in[15];
  const float* dec_b_hh = (const float*)d_in[16];
  const float* vlinin = (const float*)d_in[17];
  const float* vlinout = (const float*)d_in[18];
  const float* linin = (const float*)d_in[19];
  const float* dvlinin = (const float*)d_in[20];
  const float* linout = (const float*)d_in[21];
  const float* gen_w = (const float*)d_in[22];
  const float* gen_b = (const float*)d_in[23];
  const float* vgen_w = (const float*)d_in[24];
  const float* vgen_b = (const float*)d_in[25];
  const int* inp = (const int*)d_in[26];
  const int* verbs = (const int*)d_in[27];
  const int* outtok = (const int*)d_in[28];
  float* out = (float*)d_out;

  // Padded row counts so the logits GEMM needs no bounds checks in its K-loop:
  // nblocks(V=32000) = 63 -> 63*512 = 32256 rows; nblocks(VV=5000) = 10 -> 5120.
  const size_t VPAD = 32256, VVPAD = 5120;

  char* p = (char*)d_ws;
  bf16_t* encWih = (bf16_t*)bump(p, (size_t)2 * 2 * 1024 * 512 * 2);
  bf16_t* encWhh = (bf16_t*)bump(p, (size_t)2 * 2 * 1024 * 256 * 2);
  bf16_t* vWih0 = (bf16_t*)bump(p, (size_t)2048 * 1024 * 2);
  bf16_t* vWih1 = (bf16_t*)bump(p, (size_t)2048 * 512 * 2);
  bf16_t* vWhh = (bf16_t*)bump(p, (size_t)2 * 2048 * 512 * 2);
  bf16_t* dWih0 = (bf16_t*)bump(p, (size_t)2048 * 1024 * 2);
  bf16_t* dWih1 = (bf16_t*)bump(p, (size_t)2048 * 512 * 2);
  bf16_t* dWhh = (bf16_t*)bump(p, (size_t)2 * 2048 * 512 * 2);
  bf16_t* Wvlinin = (bf16_t*)bump(p, (size_t)512 * 512 * 2);
  bf16_t* Wvlinout = (bf16_t*)bump(p, (size_t)512 * 1024 * 2);
  bf16_t* Wlinin = (bf16_t*)bump(p, (size_t)512 * 512 * 2);
  bf16_t* Wdvlinin = (bf16_t*)bump(p, (size_t)512 * 512 * 2);
  bf16_t* Wlinout = (bf16_t*)bump(p, (size_t)512 * 1536 * 2);
  bf16_t* WgenW = (bf16_t*)bump(p, VPAD * 512 * 2);
  bf16_t* WvgenW = (bf16_t*)bump(p, VVPAD * 512 * 2);
  bf16_t* x0_bf = (bf16_t*)bump(p, (size_t)S_ * B_ * 512 * 2);
  bf16_t* hs_bf = (bf16_t*)bump(p, (size_t)S_ * B_ * 512 * 2);
  float* enc_f = (float*)bump(p, (size_t)S_ * B_ * 512 * 4);
  float* xpart = (float*)bump(p, (size_t)2 * S_ * B_ * 1024 * 4);
  float* h_state = (float*)bump(p, (size_t)2 * 16 * 512 * 4);
  float* c_state = (float*)bump(p, (size_t)2 * 16 * 512 * 4);
  float* op_state = (float*)bump(p, (size_t)16 * 512 * 4);
  float* venc_f = (float*)bump(p, (size_t)TV_ * 16 * 512 * 4);
  bf16_t* vopn_bf = (bf16_t*)bump(p, (size_t)TV_ * 16 * 512 * 2);
  bf16_t* opn_bf = (bf16_t*)bump(p, (size_t)T_ * 16 * 512 * 2);

  auto cvt = [&](const float* s, bf16_t* d, size_t n) {
    int grid = (int)((n + 2047) / 2048);
    if (grid > 4096) grid = 4096;
    if (grid < 1) grid = 1;
    cvt_kernel<<<grid, 256, 0, stream>>>(s, d, n);
  };
  cvt(enc_w_ih, encWih, (size_t)2 * 2 * 1024 * 512);
  cvt(enc_w_hh, encWhh, (size_t)2 * 2 * 1024 * 256);
  cvt(vdec_w_ih0, vWih0, (size_t)2048 * 1024);
  cvt(vdec_w_ih1, vWih1, (size_t)2048 * 512);
  cvt(vdec_w_hh, vWhh, (size_t)2 * 2048 * 512);
  cvt(dec_w_ih0, dWih0, (size_t)2048 * 1024);
  cvt(dec_w_ih1, dWih1, (size_t)2048 * 512);
  cvt(dec_w_hh, dWhh, (size_t)2 * 2048 * 512);
  cvt(vlinin, Wvlinin, (size_t)512 * 512);
  cvt(vlinout, Wvlinout, (size_t)512 * 1024);
  cvt(linin, Wlinin, (size_t)512 * 512);
  cvt(dvlinin, Wdvlinin, (size_t)512 * 512);
  cvt(linout, Wlinout, (size_t)512 * 1536);
  cvt(gen_w, WgenW, (size_t)V_ * 512);
  cvt(vgen_w, WvgenW, (size_t)VV_ * 512);
  // Zero the padded tails (columns are lane-independent in WMMA, but keep the
  // pad clean so no NaNs/denormals flow through the datapath).
  zero_bf16_kernel<<<64, 256, 0, stream>>>(WgenW + (size_t)V_ * 512,
                                           (VPAD - V_) * 512);
  zero_bf16_kernel<<<32, 256, 0, stream>>>(WvgenW + (size_t)VV_ * 512,
                                           (VVPAD - VV_) * 512);

  auto gemm = [&](const bf16_t* A, const bf16_t* W, const float* b1, const float* b2,
                  float* C, int M, int N, int K, long sB, long sT) {
    int ntiles = (N + 15) >> 4;
    int nblocks = (ntiles + 31) / 32;
    gemm_bf16_kernel<<<(M >> 4) * nblocks, 256, 0, stream>>>(A, W, b1, b2, C, M, N, K,
                                                             sB, sT);
  };

  // Encoder
  enc_embed_kernel<<<512, 256, 0, stream>>>(encemb, inp, x0_bf);
  const size_t encSmem = 16 * 1024 * 4 + 16 * 256 * 4 + 16 * 256 * 2;
  for (int l = 0; l < 2; ++l) {
    const bf16_t* Ain = (l == 0) ? x0_bf : hs_bf;
    for (int d = 0; d < 2; ++d) {
      gemm(Ain, encWih + (size_t)(l * 2 + d) * 1024 * 512,
           enc_b_ih + (size_t)(l * 2 + d) * 1024, enc_b_hh + (size_t)(l * 2 + d) * 1024,
           xpart + (size_t)d * S_ * B_ * 1024, S_ * B_, 1024, 512, 1024L, 16 * 1024L);
    }
    enc_lstm_kernel<<<2, 256, encSmem, stream>>>(
        xpart, encWhh + (size_t)l * 2 * 1024 * 256, (l == 1) ? enc_f : nullptr,
        (l == 0) ? hs_bf : nullptr, h_state + (size_t)l * 16 * 512,
        c_state + (size_t)l * 16 * 512);
  }

  // Decoders (sequential recurrences; logits hoisted to big GEMMs after)
  zero_kernel<<<8, 256, 0, stream>>>(op_state, (size_t)16 * 512);
  const size_t vSmem = 131072 + 32768 + 16384 + 16384 + 8192;          // 204800
  vdec_kernel<<<1, 256, vSmem, stream>>>(
      verbs, vemb, vWih0, vWih1, vWhh, vWhh + (size_t)2048 * 512, vdec_b_ih, vdec_b_hh,
      Wvlinin, Wvlinout, enc_f, h_state, c_state, op_state, venc_f, vopn_bf);
  const size_t dSmem = 131072 + 32768 + 16384 + 16384 + 49152 + 8192 + 2048; // 256000
  dec_kernel<<<1, 256, dSmem, stream>>>(
      outtok, decemb, dWih0, dWih1, dWhh, dWhh + (size_t)2048 * 512, dec_b_ih, dec_b_hh,
      Wlinin, Wdvlinin, Wlinout, enc_f, venc_f, h_state, c_state, op_state, opn_bf);

  // Output projections (dominant FLOPs): outputs then voutputs into d_out.
  gemm(opn_bf, WgenW, gen_b, nullptr, out, T_ * 16, V_, 512, (long)T_ * V_, (long)V_);
  gemm(vopn_bf, WvgenW, vgen_b, nullptr, out + (size_t)B_ * T_ * V_, TV_ * 16, VV_, 512,
       (long)TV_ * VV_, (long)VV_);
}
